// My_super_loss_kmeans2_52725018526337
// MI455X (gfx1250) — matrix operand order repaired and checked by reading, not compiled, not run on hardware
//
#include <hip/hip_runtime.h>

typedef __attribute__((ext_vector_type(16))) _Float16 v16h;
typedef __attribute__((ext_vector_type(8)))  _Float16 v8h;
typedef __attribute__((ext_vector_type(8)))  float    v8f;

#define BB      2
#define NN      32768
#define CC      256
#define DS      2048          // NN / 16
#define KC      5
#define MAXIT   100
#define TOLER   1e-3f
#define NT      (DS / 16)     // 128 row-tiles
#define TPC     32            // tiles per LDS-dots chunk

// workspace layout (float units) per problem p in 0..3
#define PSTR    (1u<<20)      // 4 MB per problem
#define F32OFF  0u            // fea32  [DS*CC]
#define F16OFF  524288u       // fea16  [DS*CC] halfs (262144 float slots)
#define POSOFF  786432u       // pos3   [DS*3]
#define IDXOFF  792576u       // idx_ds [DS] ints
#define C3OFF   794624u       // cent3  [KC*3]

// ---------------------------------------------------------------- utilities
__device__ __forceinline__ int block_argmax(const float* v, int n,
                                            float* redv, int* redi) {
  int tid = threadIdx.x;
  float bv = -3.0e38f; int bi = 0;
  for (int r = tid; r < n; r += 256)
    if (v[r] > bv) { bv = v[r]; bi = r; }
  redv[tid] = bv; redi[tid] = bi;
  __syncthreads();
  for (int s = 128; s > 0; s >>= 1) {
    if (tid < s) {
      if (redv[tid + s] > redv[tid] ||
          (redv[tid + s] == redv[tid] && redi[tid + s] < redi[tid])) {
        redv[tid] = redv[tid + s]; redi[tid] = redi[tid + s];
      }
    }
    __syncthreads();
  }
  int far = redi[0];
  __syncthreads();
  return far;
}

// ---------------------------------------------------------------- kernels
__global__ void k_zero(float* out) { out[0] = 0.f; }

// cosine-similarity loss: one wave32 per 256-dim row
__global__ void k_cosloss(const float* __restrict__ a,
                          const float* __restrict__ b, float* out) {
  int row  = blockIdx.x * 8 + (threadIdx.x >> 5);
  int lane = threadIdx.x & 31;
  const float* ar = a + (size_t)row * CC;
  const float* br = b + (size_t)row * CC;
  float num = 0.f, na = 0.f, nb = 0.f;
  #pragma unroll
  for (int i = lane; i < CC; i += 32) {
    float x = ar[i], y = br[i];
    num += x * y; na += x * x; nb += y * y;
  }
  #pragma unroll
  for (int off = 16; off >= 1; off >>= 1) {
    num += __shfl_xor(num, off, 32);
    na  += __shfl_xor(na,  off, 32);
    nb  += __shfl_xor(nb,  off, 32);
  }
  if (lane == 0) {
    float c = num / fmaxf(sqrtf(na) * sqrtf(nb), 1e-8f);
    atomicAdd(out, -c / (float)(BB * NN));
  }
}

// furthest-point sampling: one persistent 1024-thread block per problem
__global__ void __launch_bounds__(1024)
k_fps(const float* __restrict__ p0first, const float* __restrict__ p0sec,
      float* ws) {
  int p = blockIdx.x;
  const float* pos = ((p < 2) ? p0first : p0sec) + (size_t)(p & 1) * NN * 3;
  int* idx_ds = (int*)(ws + (size_t)p * PSTR + IDXOFF);

  float px[32], py[32], pz[32], dmin[32];
  #pragma unroll
  for (int t = 0; t < 32; ++t) {
    int r = threadIdx.x + t * 1024;
    px[t] = pos[r * 3 + 0]; py[t] = pos[r * 3 + 1]; pz[t] = pos[r * 3 + 2];
    dmin[t] = 1e10f;
  }
  __shared__ float lv[32]; __shared__ int li[32]; __shared__ int scur;
  int cur = 0;
  for (int s = 0; s < DS; ++s) {
    if (threadIdx.x == 0) idx_ds[s] = cur;
    float lx = pos[cur * 3], ly = pos[cur * 3 + 1], lz = pos[cur * 3 + 2];
    float bv = -1.f; int bi = 0;
    #pragma unroll
    for (int t = 0; t < 32; ++t) {
      float dx = px[t] - lx, dy = py[t] - ly, dz = pz[t] - lz;
      float d = dx * dx + dy * dy + dz * dz;
      dmin[t] = fminf(dmin[t], d);
      if (dmin[t] > bv) { bv = dmin[t]; bi = threadIdx.x + t * 1024; }
    }
    #pragma unroll
    for (int off = 16; off >= 1; off >>= 1) {
      float ov = __shfl_xor(bv, off, 32); int oi = __shfl_xor(bi, off, 32);
      if (ov > bv || (ov == bv && oi < bi)) { bv = ov; bi = oi; }
    }
    if ((threadIdx.x & 31) == 0) { lv[threadIdx.x >> 5] = bv; li[threadIdx.x >> 5] = bi; }
    __syncthreads();
    if (threadIdx.x < 32) {
      bv = lv[threadIdx.x]; bi = li[threadIdx.x];
      #pragma unroll
      for (int off = 16; off >= 1; off >>= 1) {
        float ov = __shfl_xor(bv, off, 32); int oi = __shfl_xor(bi, off, 32);
        if (ov > bv || (ov == bv && oi < bi)) { bv = ov; bi = oi; }
      }
      if (threadIdx.x == 0) scur = bi;
    }
    __syncthreads();
    cur = scur;
    __syncthreads();
  }
}

// gather downsampled rows; produce f32 + f16 copies of features
__global__ void k_gather(const float* __restrict__ logits,
                         const float* __restrict__ logits1,
                         const float* __restrict__ p0first,
                         const float* __restrict__ p0sec, float* ws) {
  int p = blockIdx.x >> 11;
  int r = blockIdx.x & (DS - 1);
  float*     base  = ws + (size_t)p * PSTR;
  const int* idxds = (const int*)(base + IDXOFF);
  int src = idxds[r];
  const float* fsrc = ((p < 2) ? logits : logits1) +
                      ((size_t)(p & 1) * NN + src) * CC;
  const float* psrc = ((p < 2) ? p0first : p0sec) +
                      ((size_t)(p & 1) * NN + src) * 3;
  float v = fsrc[threadIdx.x];
  base[F32OFF + (size_t)r * CC + threadIdx.x] = v;
  ((_Float16*)(base + F16OFF))[(size_t)r * CC + threadIdx.x] = (_Float16)v;
  if (threadIdx.x < 3) base[POSOFF + r * 3 + threadIdx.x] = psrc[threadIdx.x];
}

// greedy initial centroids on the 3-D positions
__global__ void k_initc(float* ws) {
  int p = blockIdx.x, tid = threadIdx.x;
  float* base = ws + (size_t)p * PSTR;
  const float* pos3 = base + POSOFF;
  __shared__ float sp[DS * 3];
  __shared__ float sc[KC * 3];
  __shared__ float redv[256]; __shared__ int redi[256];
  for (int i = tid; i < DS * 3; i += 256) sp[i] = pos3[i];
  __syncthreads();
  if (tid < 3) sc[tid] = sp[tid];
  __syncthreads();
  for (int k = 1; k < KC; ++k) {
    float bv = -1.f; int bi = 0;
    for (int r = tid; r < DS; r += 256) {
      float d2 = 0.f;
      for (int j = 0; j < k; ++j) {
        float dx = sp[r * 3] - sc[j * 3], dy = sp[r * 3 + 1] - sc[j * 3 + 1],
              dz = sp[r * 3 + 2] - sc[j * 3 + 2];
        d2 += dx * dx + dy * dy + dz * dz;
      }
      if (d2 > bv) { bv = d2; bi = r; }
    }
    redv[tid] = bv; redi[tid] = bi;
    __syncthreads();
    for (int s = 128; s > 0; s >>= 1) {
      if (tid < s) {
        if (redv[tid + s] > redv[tid] ||
            (redv[tid + s] == redv[tid] && redi[tid + s] < redi[tid])) {
          redv[tid] = redv[tid + s]; redi[tid] = redi[tid + s];
        }
      }
      __syncthreads();
    }
    if (tid == 0) {
      int r = redi[0];
      sc[k * 3] = sp[r * 3]; sc[k * 3 + 1] = sp[r * 3 + 1]; sc[k * 3 + 2] = sp[r * 3 + 2];
    }
    __syncthreads();
  }
  if (tid < KC * 3) base[C3OFF + tid] = sc[tid];
}

// ------------------------------------------------ persistent k-means kernel
__global__ void __launch_bounds__(256)
k_kmeans(float* ws, float* out) {
  int tid = threadIdx.x, p = blockIdx.x;
  float*     base  = ws + (size_t)p * PSTR;
  const float*     fea32 = base + F32OFF;
  const _Float16*  fea16 = (const _Float16*)(base + F16OFF);
  const float*     pos3  = base + POSOFF;

  __shared__ float    scent[KC * CC];
  __shared__ float    sacc[KC * CC];
  __shared__ _Float16 scent16[16 * CC];   // zero-padded f16 centroids (WMMA B)
  __shared__ float    sdots[TPC * 256];   // per-chunk dot tiles: [lt][col*16+rr]
  __shared__ float    scent3[KC * 3];
  __shared__ float    scount[KC];
  __shared__ float    cnorm[KC];
  __shared__ int      sidx[DS];
  __shared__ float    smind[DS];
  __shared__ float    sxnorm[DS];
  __shared__ float    redv[256]; __shared__ int redi[256];
  __shared__ float    sscal[4];   // 0:sse_pre 1:sse_fin 2:sse_n 3:accum
  __shared__ int      sdone;

  if (tid < KC * 3) scent3[tid] = base[C3OFF + tid];
  // per-row squared norms of the 256-d features (vectorized b128 loads)
  for (int r = tid; r < DS; r += 256) {
    const float4* xr = (const float4*)(fea32 + (size_t)r * CC);
    float s = 0.f;
    #pragma unroll 4
    for (int d = 0; d < CC / 4; ++d) {
      float4 v = xr[d];
      s += v.x * v.x + v.y * v.y + v.z * v.z + v.w * v.w;
    }
    sxnorm[r] = s;
  }
  __syncthreads();

  // ---------- helpers inlined as lambdas ----------
  auto assign3 = [&](const float* X, int xs) {
    float cx[KC], cy[KC], cz[KC];
    #pragma unroll
    for (int k = 0; k < KC; ++k) {
      cx[k] = scent3[k * 3]; cy[k] = scent3[k * 3 + 1]; cz[k] = scent3[k * 3 + 2];
    }
    for (int r = tid; r < DS; r += 256) {
      const float* xr = X + (size_t)r * xs;
      float x0 = xr[0], x1 = xr[1], x2 = xr[2];
      float best = 3.0e38f; int bk = 0;
      #pragma unroll
      for (int k = 0; k < KC; ++k) {
        float dx = x0 - cx[k], dy = x1 - cy[k], dz = x2 - cz[k];
        float d = dx * dx + dy * dy + dz * dz;
        if (d < best) { best = d; bk = k; }
      }
      sidx[r] = bk; smind[r] = best;
    }
    __syncthreads();
  };

  auto sse3 = [&](const float* X, int xs, const float* c, int cs) -> float {
    if (tid == 0) sscal[3] = 0.f;
    __syncthreads();
    float cx[KC], cy[KC], cz[KC];
    #pragma unroll
    for (int k = 0; k < KC; ++k) {
      cx[k] = c[k * cs]; cy[k] = c[k * cs + 1]; cz[k] = c[k * cs + 2];
    }
    float loc = 0.f;
    for (int r = tid; r < DS; r += 256) {
      int k = sidx[r];
      const float* xr = X + (size_t)r * xs;
      float dx = xr[0] - cx[k], dy = xr[1] - cy[k], dz = xr[2] - cz[k];
      loc += dx * dx + dy * dy + dz * dz;
    }
    atomicAdd(&sscal[3], loc);
    __syncthreads();
    return sqrtf(sscal[3]);
  };

  auto recompute3 = [&](const float* X, int xs) {
    __shared__ float sacc3[KC * 3];
    if (tid < 3) {
      float r0 = 0, r1 = 0, r2 = 0, r3 = 0, r4 = 0;
      for (int r = 0; r < DS; ++r) {
        float v = X[(size_t)r * xs + tid];
        int k = sidx[r];
        r0 += (k == 0) ? v : 0.f; r1 += (k == 1) ? v : 0.f;
        r2 += (k == 2) ? v : 0.f; r3 += (k == 3) ? v : 0.f;
        r4 += (k == 4) ? v : 0.f;
      }
      sacc3[0 * 3 + tid] = r0; sacc3[1 * 3 + tid] = r1; sacc3[2 * 3 + tid] = r2;
      sacc3[3 * 3 + tid] = r3; sacc3[4 * 3 + tid] = r4;
    }
    if (tid < KC) {
      float c = 0.f;
      for (int r = 0; r < DS; ++r) c += (sidx[r] == tid) ? 1.f : 0.f;
      scount[tid] = c;
    }
    __syncthreads();
    if (tid < KC * 3) scent3[tid] = sacc3[tid] / fmaxf(scount[tid / 3], 1.f);
    __syncthreads();
    int far = block_argmax(smind, DS, redv, redi);
    if (tid < KC * 3 && scount[tid / 3] == 0.f)
      scent3[tid] = X[(size_t)far * xs + (tid % 3)];
    __syncthreads();
  };

  auto recompute_full = [&]() {
    { // dimension-major accumulation: thread owns dim d = tid, no atomics
      float r0 = 0, r1 = 0, r2 = 0, r3 = 0, r4 = 0;
      for (int r = 0; r < DS; ++r) {
        float v = fea32[(size_t)r * CC + tid];
        int k = sidx[r];
        r0 += (k == 0) ? v : 0.f; r1 += (k == 1) ? v : 0.f;
        r2 += (k == 2) ? v : 0.f; r3 += (k == 3) ? v : 0.f;
        r4 += (k == 4) ? v : 0.f;
      }
      sacc[0 * CC + tid] = r0; sacc[1 * CC + tid] = r1; sacc[2 * CC + tid] = r2;
      sacc[3 * CC + tid] = r3; sacc[4 * CC + tid] = r4;
    }
    if (tid < KC) {
      float c = 0.f;
      for (int r = 0; r < DS; ++r) c += (sidx[r] == tid) ? 1.f : 0.f;
      scount[tid] = c;
    }
    __syncthreads();
    for (int i = tid; i < KC * CC; i += 256)
      scent[i] = sacc[i] / fmaxf(scount[i / CC], 1.f);
    __syncthreads();
    int far = block_argmax(smind, DS, redv, redi);
    for (int i = tid; i < KC * CC; i += 256)
      if (scount[i / CC] == 0.f) scent[i] = fea32[(size_t)far * CC + (i % CC)];
    __syncthreads();
    if (tid < KC) {
      float s = 0.f;
      for (int d = 0; d < CC; ++d) { float v = scent[tid * CC + d]; s += v * v; }
      cnorm[tid] = s;
    }
    // zero-padded f16 copy for the WMMA B operand (rows >= KC are zero)
    for (int i = tid; i < 16 * CC; i += 256) {
      int k = i / CC;
      scent16[i] = (k < KC) ? (_Float16)scent[i] : (_Float16)0.f;
    }
    __syncthreads();
  };

  // WMMA assignment: argmin_k (|c_k|^2 - 2 x·c_k), x·c via v_wmma f16->f32.
  // Dots are dumped contiguously to LDS; a separate pass does the argmin, so
  // the hot loop is pure (b128 loads + wmma + 2 b128 stores), no shuffles.
  auto assign_wmma = [&]() {
    int lane = tid & 31, wave = tid >> 5;
    int col = lane & 15, half = lane >> 4;
    // hoist all 8 B fragments into registers (invariant over tiles)
    v16h bfrag[8];
    #pragma unroll
    for (int kb = 0; kb < CC / 32; ++kb)
      bfrag[kb] = __builtin_shufflevector(
          *(const v8h*)&scent16[col * CC + kb * 32 + half * 8],
          *(const v8h*)&scent16[col * CC + kb * 32 + 16 + half * 8],
          0, 1, 2, 3, 4, 5, 6, 7, 8, 9, 10, 11, 12, 13, 14, 15);
    float cn[KC];
    #pragma unroll
    for (int k = 0; k < KC; ++k) cn[k] = cnorm[k];

    for (int c = 0; c < NT / TPC; ++c) {
      for (int t = c * TPC + wave; t < (c + 1) * TPC; t += 8) {
        int arow = t * 16 + col;
        const _Float16* fr = fea16 + (size_t)arow * CC;
        if (t + 8 < NT)                 // cover next tile's A latency
          __builtin_prefetch(fr + (size_t)128 * CC, 0, 1);
        v8f acc = {};
        #pragma unroll
        for (int kb = 0; kb < CC / 32; ++kb) {
          v8h lo = *(const v8h*)(fr + kb * 32 + half * 8);
          v8h hi = *(const v8h*)(fr + kb * 32 + 16 + half * 8);
          v16h a = __builtin_shufflevector(lo, hi, 0, 1, 2, 3, 4, 5, 6, 7,
                                                   8, 9, 10, 11, 12, 13, 14, 15);
          acc = __builtin_amdgcn_wmma_f32_16x16x32_f16(
              false, a, false, bfrag[kb], (short)0, acc, false, false);
        }
        // lane holds rows (half*8 + 0..7) of column `col`: 32B contiguous
        float* dp = &sdots[(t - c * TPC) * 256 + col * 16 + half * 8];
        *(float4*)dp       = make_float4(acc[0], acc[1], acc[2], acc[3]);
        *(float4*)(dp + 4) = make_float4(acc[4], acc[5], acc[6], acc[7]);
      }
      __syncthreads();
      // argmin pass over this chunk's 512 rows (2 rows/thread)
      for (int ro = c * (TPC * 16) + tid; ro < (c + 1) * (TPC * 16); ro += 256) {
        int lt = (ro >> 4) - c * TPC, rr = ro & 15;
        const float* dr = &sdots[lt * 256 + rr];
        float best = 3.0e38f; int bk = 0;
        #pragma unroll
        for (int k = 0; k < KC; ++k) {
          float sc = cn[k] - 2.0f * dr[k * 16];
          if (sc < best) { best = sc; bk = k; }
        }
        sidx[ro] = bk; smind[ro] = best + sxnorm[ro];
      }
      __syncthreads();
    }
  };

  // ---------------- phase A: positions ----------------
  assign3(pos3, 3);
  recompute3(pos3, 3);
  float sse_n = sse3(pos3, 3, scent3, 3);
  if (tid == 0) sscal[2] = sse_n;

  // ---------------- phase B: iteration 0 (fea[:, :3] vs 3-D centroids) -----
  assign3(fea32, CC);
  recompute_full();
  float s0 = sse3(fea32, CC, scent, CC);
  if (tid == 0) {
    sscal[0] = s0; sscal[1] = s0;
    sdone = (fabsf(s0) < TOLER) ? 1 : 0;
  }
  __syncthreads();

  // ---------------- phase C: masked iterations with WMMA assignment -------
  for (int it = 1; it < MAXIT; ++it) {
    if (sdone) break;                 // uniform across block
    assign_wmma();
    recompute_full();
    float s2 = sse3(fea32, CC, scent, CC);
    if (tid == 0) {
      if (fabsf(s2 - sscal[0]) < TOLER) sdone = 1;
      sscal[1] = s2; sscal[0] = s2;
    }
    __syncthreads();
  }
  if (tid == 0) atomicAdd(out, sscal[2] + sscal[1]);
}

// ---------------------------------------------------------------- launcher
extern "C" void kernel_launch(void* const* d_in, const int* in_sizes, int n_in,
                              void* d_out, int out_size, void* d_ws, size_t ws_size,
                              hipStream_t stream) {
  const float* logits  = (const float*)d_in[0];
  const float* logits1 = (const float*)d_in[1];
  const float* p0first = (const float*)d_in[2];
  const float* p0sec   = (const float*)d_in[3];
  float* out = (float*)d_out;
  float* ws  = (float*)d_ws;
  (void)in_sizes; (void)n_in; (void)out_size; (void)ws_size;

  k_zero<<<1, 1, 0, stream>>>(out);
  k_cosloss<<<(BB * NN) / 8, 256, 0, stream>>>(logits, logits1, out);
  k_fps<<<4, 1024, 0, stream>>>(p0first, p0sec, ws);
  k_gather<<<4 * DS, 256, 0, stream>>>(logits, logits1, p0first, p0sec, ws);
  k_initc<<<4, 256, 0, stream>>>(ws);
  k_kmeans<<<4, 256, 0, stream>>>(ws, out);
}
